// WindowAttention_75084618268794
// MI455X (gfx1250) — matrix operand order repaired and checked by reading, not compiled, not run on hardware
//
#include <hip/hip_runtime.h>

typedef _Float16 half_t;
typedef _Float16 v16h __attribute__((ext_vector_type(16)));
typedef float    v8f  __attribute__((ext_vector_type(8)));

#define BLOCK 256
#define WAVES 8

// ---- problem constants ----
#define NWIN 1024
#define NTOK 64
#define DIM 512
#define HEADS 16
#define HDIM 32
#define MAX_LOGIT 4.605170185988091f   // log(100)

// ---- LDS layout (element strides, padded to spread banks) ----
#define XS_LD   520    // x tile f16 rows (row stride 1040B -> conflict-free A frags)
#define TMPF_LD 33     // q/k/v f32 scratch rows
#define QS_LD   40     // normalized q/k f16 rows
#define VT_LD   72     // v-transposed f16 rows
#define SF_LD   66     // softmax scores f32 rows
#define PS_LD   72     // probs f16 rows

// byte offsets within dynamic LDS
#define XS_OFF    0
#define TMPF_OFF  (XS_OFF   + 64 * XS_LD * 2)          // 66560
#define QS_OFF    (TMPF_OFF + 3 * 64 * TMPF_LD * 4)    // 91904
#define KS_OFF    (QS_OFF   + 64 * QS_LD * 2)          // 97024
#define VT_OFF    (KS_OFF   + 64 * QS_LD * 2)          // 102144
#define SF_OFF    (VT_OFF   + 32 * VT_LD * 2)          // 106752 (aliased as o16)
#define PS_OFF    (SF_OFF   + 64 * SF_LD * 4)          // 123648
#define RINV_OFF  (PS_OFF   + 64 * PS_LD * 2)          // 132864
#define SMEM_BYTES (RINV_OFF + 128 * 4)                // 133376  (2 blocks/WGP)

// ---- workspace layout (bytes) ----
#define WS_QKVW   0u            // 786432 f16
#define WS_PROJW  1572864u      // 262144 f16
#define WS_SCALES 2097152u      // 16 f32
#define WS_BIASC  4194304u      // 64*16*4096 f32 = 16 MiB (mask + rpb combined)
#define WS_AO     20971520u     // 1024*64*512 f16 = 64 MiB

// ---------------------------------------------------------------------------
// Fragment loaders per CDNA5 ISA 7.12.2 (wave32, 16-bit operands)
// A (16x32, row-major src): lane&15 = M row; lanes<16 hold K {k0..k0+7, k0+16..k0+23},
// lanes>=16 hold K {k0+8..k0+15, k0+24..k0+31}.
// ---------------------------------------------------------------------------
__device__ __forceinline__ v16h load_frag_a(const half_t* __restrict__ s, int ld,
                                            int m0, int k0, int lane) {
  const half_t* p = s + (m0 + (lane & 15)) * ld + k0 + ((lane >> 4) << 3);
  v16h a;
#pragma unroll
  for (int j = 0; j < 8; ++j) { a[j] = p[j]; a[j + 8] = p[16 + j]; }
  return a;
}

// B (32x16) where B[k][n] = W[n][k] (W row-major): lane&15 = N col;
// lanes<16 hold K {k0..k0+7, k0+16..k0+23}, lanes>=16 hold K {k0+8..k0+15, k0+24..k0+31}.
__device__ __forceinline__ v16h load_frag_bt(const half_t* __restrict__ w, int ld,
                                             int n0, int k0, int lane) {
  const half_t* p = w + (n0 + (lane & 15)) * ld + k0 + ((lane >> 4) << 3);
  v16h b;
#pragma unroll
  for (int j = 0; j < 8; ++j) { b[j] = p[j]; b[j + 8] = p[16 + j]; }
  return b;
}

#define WMMA_F16(a, b, c) \
  __builtin_amdgcn_wmma_f32_16x16x32_f16(false, (a), false, (b), (short)0, (c), false, false)

// ---------------------------------------------------------------------------
// Prep: f32->f16 weights, combined (mask + rpb) bias table, per-head scales
// ---------------------------------------------------------------------------
__global__ void prep_kernel(const float* __restrict__ qkv_w,
                            const float* __restrict__ proj_w,
                            const float* __restrict__ mask,
                            const float* __restrict__ rpb_table,
                            const int*   __restrict__ rel_index,
                            const float* __restrict__ logit_scale,
                            half_t* __restrict__ qkvw_h,
                            half_t* __restrict__ projw_h,
                            float*  __restrict__ biasc,
                            float*  __restrict__ scales) {
  const unsigned NQ = 3u * DIM * DIM;                 // 786432
  const unsigned NP = DIM * DIM;                      // 262144
  const unsigned NB = 64u * HEADS * NTOK * NTOK;      // 4194304
  const unsigned total = NQ + NP + NB + HEADS;
  for (unsigned i = blockIdx.x * blockDim.x + threadIdx.x; i < total;
       i += gridDim.x * blockDim.x) {
    if (i < NQ) {
      qkvw_h[i] = (half_t)qkv_w[i];
    } else if (i < NQ + NP) {
      unsigned j = i - NQ;
      projw_h[j] = (half_t)proj_w[j];
    } else if (i < NQ + NP + NB) {
      unsigned j = i - NQ - NP;
      unsigned wi = j >> 16;                 // window-in-image 0..63
      unsigned rem = j & 65535u;
      unsigned h = rem >> 12, p = rem & 4095u;
      biasc[j] = mask[wi * 4096u + p] + rpb_table[rel_index[p] * HEADS + h];
    } else {
      unsigned h = i - NQ - NP - NB;
      scales[h] = __expf(fminf(logit_scale[h], MAX_LOGIT));
    }
  }
}

// ---------------------------------------------------------------------------
// Fused per-window attention: one block per window (1024 blocks, 8 wave32)
// ---------------------------------------------------------------------------
__global__ __launch_bounds__(BLOCK)
void swin_attn_kernel(const float* __restrict__ x,
                      const half_t* __restrict__ qkvw,
                      const float* __restrict__ biasc,
                      const float* __restrict__ scales,
                      half_t* __restrict__ ao) {
  const int w    = blockIdx.x;
  const int tid  = threadIdx.x;
  const int lane = tid & 31;
  const int wave = tid >> 5;

  extern __shared__ __align__(16) char smem[];
  half_t* xs   = (half_t*)(smem + XS_OFF);
  float*  tmpf = (float*)(smem + TMPF_OFF);
  half_t* qs   = (half_t*)(smem + QS_OFF);
  half_t* ks   = (half_t*)(smem + KS_OFF);
  half_t* vt   = (half_t*)(smem + VT_OFF);
  float*  sf   = (float*)(smem + SF_OFF);
  half_t* o16  = (half_t*)(smem + SF_OFF);    // alias: O tile bounce (64x32 f16)
  half_t* ps   = (half_t*)(smem + PS_OFF);
  float*  rinv = (float*)(smem + RINV_OFF);   // [0..63] q, [64..127] k

  // Stage x tile as f16 (64x512), reused by all 16 heads.
  const float* xp = x + (size_t)w * NTOK * DIM;
#pragma unroll 4
  for (int i = tid; i < NTOK * DIM; i += BLOCK) {
    int r = i >> 9, c = i & 511;
    xs[r * XS_LD + c] = (half_t)xp[i];
  }
  __syncthreads();

  for (int h = 0; h < HEADS; ++h) {
    // Warm next head's weight rows into cache (global_prefetch_b8).
    if (h + 1 < HEADS && wave < 3) {
      const half_t* nw =
          qkvw + ((size_t)(wave * DIM + (h + 1) * HDIM + lane)) * DIM;
      __builtin_prefetch(nw, 0, 0);
    }

    // ---- QKV GEMM for this head: 3 GEMMs of (64x32, K=512) -> 24 tiles ----
    for (int t = wave; t < 24; t += WAVES) {
      int which = t >> 3;                 // 0=q 1=k 2=v
      int idx = t & 7;
      int mi = idx >> 1, ni = idx & 1;
      const half_t* wp =
          qkvw + ((size_t)(which * DIM + h * HDIM + ni * 16)) * DIM;
      v8f acc = {};
#pragma unroll 4
      for (int kk = 0; kk < 16; ++kk) {
        v16h a = load_frag_a(xs, XS_LD, mi * 16, kk * 32, lane);
        v16h b = load_frag_bt(wp, DIM, 0, kk * 32, lane);
        acc = WMMA_F16(a, b, acc);
      }
      float* dst = tmpf + which * (64 * TMPF_LD);
      int n  = ni * 16 + (lane & 15);
      int mb = mi * 16 + ((lane >> 4) << 3);
#pragma unroll
      for (int r = 0; r < 8; ++r) dst[(mb + r) * TMPF_LD + n] = acc[r];
    }
    __syncthreads();

    // ---- q/k row L2 norms ----
    if (tid < 128) {
      int which = tid >> 6, r = tid & 63;
      const float* src = tmpf + which * (64 * TMPF_LD) + r * TMPF_LD;
      float ss = 0.f;
#pragma unroll
      for (int d = 0; d < HDIM; ++d) ss += src[d] * src[d];
      rinv[tid] = 1.0f / fmaxf(sqrtf(ss), 1e-12f);
    }
    __syncthreads();

    // ---- normalize -> f16 qs/ks; v -> transposed f16 vt ----
    for (int i = tid; i < NTOK * HDIM; i += BLOCK) {
      int r = i >> 5, d = i & 31;
      qs[r * QS_LD + d] = (half_t)(tmpf[r * TMPF_LD + d] * rinv[r]);
      ks[r * QS_LD + d] =
          (half_t)(tmpf[64 * TMPF_LD + r * TMPF_LD + d] * rinv[64 + r]);
      vt[d * VT_LD + r] = (half_t)(tmpf[2 * 64 * TMPF_LD + r * TMPF_LD + d]);
    }
    __syncthreads();

    // ---- S = (qn . kn) * scale + bias   (16 tiles, K=32 one-shot) ----
    const float scale = scales[h];
    const float* biasp =
        biasc + ((size_t)(w & 63) * HEADS + h) * (NTOK * NTOK);
    for (int t = wave; t < 16; t += WAVES) {
      int mi = t >> 2, ni = t & 3;
      v16h a = load_frag_a(qs, QS_LD, mi * 16, 0, lane);
      v16h b = load_frag_bt(ks, QS_LD, ni * 16, 0, lane);
      v8f acc = {};
      acc = WMMA_F16(a, b, acc);
      int n  = ni * 16 + (lane & 15);
      int mb = mi * 16 + ((lane >> 4) << 3);
#pragma unroll
      for (int r = 0; r < 8; ++r) {
        int m = mb + r;
        sf[m * SF_LD + n] = acc[r] * scale + biasp[m * NTOK + n];
      }
    }
    __syncthreads();

    // ---- row softmax -> f16 probs ----
    if (tid < NTOK) {
      float* row = sf + tid * SF_LD;
      float mx = row[0];
#pragma unroll
      for (int c = 1; c < NTOK; ++c) mx = fmaxf(mx, row[c]);
      float sum = 0.f;
#pragma unroll
      for (int c = 0; c < NTOK; ++c) {
        float e = __expf(row[c] - mx);
        row[c] = e;
        sum += e;
      }
      float inv = 1.0f / sum;
      half_t* prow = ps + tid * PS_LD;
#pragma unroll
      for (int c = 0; c < NTOK; ++c) prow[c] = (half_t)(row[c] * inv);
    }
    __syncthreads();

    // ---- O = P @ V  (64x32, K=64 -> 8 tiles, one per wave) ----
    // sf region is dead now; reuse it as f16 O bounce buffer (64x32, ld=32).
    {
      int mi = wave >> 1, ni = wave & 1;
      v8f acc = {};
#pragma unroll
      for (int kk = 0; kk < 2; ++kk) {
        v16h a = load_frag_a(ps, PS_LD, mi * 16, kk * 32, lane);
        v16h b = load_frag_bt(vt, VT_LD, ni * 16, kk * 32, lane);
        acc = WMMA_F16(a, b, acc);
      }
      int n  = ni * 16 + (lane & 15);
      int mb = mi * 16 + ((lane >> 4) << 3);
#pragma unroll
      for (int r = 0; r < 8; ++r) o16[(mb + r) * HDIM + n] = (half_t)acc[r];
    }
    __syncthreads();

    // ---- coalesced O store: 16B per thread, 64B contiguous per token row ----
    {
      int row = tid >> 2, seg = tid & 3;        // 64 rows x 4 segments of 8 f16
      const uint4* src = (const uint4*)(o16 + row * HDIM);
      uint4 val = src[seg];
      half_t* aop = ao + (size_t)w * NTOK * DIM + (size_t)row * DIM + h * HDIM;
      *(uint4*)(aop + seg * 8) = val;
    }
    __syncthreads();   // protect tmpf/qs/ks/vt/sf/ps reuse next head
  }
}

// ---------------------------------------------------------------------------
// Output projection: out(65536x512) = ao @ proj_w^T + b, WMMA f16->f32,
// LDS-staged f32 tile for fully coalesced 256B-per-row stores.
// grid (1024 Mtiles, 8 Ntiles), block 256
// ---------------------------------------------------------------------------
#define OT_LD 68
__global__ __launch_bounds__(BLOCK)
void proj_kernel(const half_t* __restrict__ ao,
                 const half_t* __restrict__ pw,
                 const float* __restrict__ pb,
                 float* __restrict__ out) {
  __shared__ float ot[64 * OT_LD];   // 17408 B, padded: conflict-free r/w
  const int lane = threadIdx.x & 31;
  const int wave = threadIdx.x >> 5;
  const half_t* abase = ao + (size_t)blockIdx.x * 64 * DIM;
  const half_t* bbase = pw + (size_t)blockIdx.y * 64 * DIM;

  for (int t = wave; t < 16; t += WAVES) {
    int mi = t >> 2, ni = t & 3;
    v8f acc = {};
#pragma unroll 4
    for (int kk = 0; kk < 16; ++kk) {
      v16h a = load_frag_a(abase, DIM, mi * 16, kk * 32, lane);
      v16h b = load_frag_bt(bbase, DIM, ni * 16, kk * 32, lane);
      acc = WMMA_F16(a, b, acc);
    }
    int nloc = ni * 16 + (lane & 15);
    int mb   = mi * 16 + ((lane >> 4) << 3);
#pragma unroll
    for (int r = 0; r < 8; ++r) ot[(mb + r) * OT_LD + nloc] = acc[r];
  }
  __syncthreads();

  // Copy out with bias: each thread handles 16 f32 (4 x float4), rows get
  // 256B contiguous stores.
  {
    int row = threadIdx.x >> 2, seg = threadIdx.x & 3;
    const float* src = ot + row * OT_LD + seg * 16;
    int colbase = blockIdx.y * 64 + seg * 16;
    float* dst = out + ((size_t)blockIdx.x * 64 + row) * DIM + colbase;
#pragma unroll
    for (int j = 0; j < 4; ++j) {
      float4 v = *(const float4*)(src + j * 4);
      v.x += pb[colbase + j * 4 + 0];
      v.y += pb[colbase + j * 4 + 1];
      v.z += pb[colbase + j * 4 + 2];
      v.w += pb[colbase + j * 4 + 3];
      *(float4*)(dst + j * 4) = v;
    }
  }
}

// ---------------------------------------------------------------------------
extern "C" void kernel_launch(void* const* d_in, const int* in_sizes, int n_in,
                              void* d_out, int out_size, void* d_ws,
                              size_t ws_size, hipStream_t stream) {
  (void)in_sizes; (void)n_in; (void)out_size; (void)ws_size;
  const float* x           = (const float*)d_in[0];
  const float* mask        = (const float*)d_in[1];
  const float* qkv_w       = (const float*)d_in[2];
  const float* proj_w      = (const float*)d_in[3];
  const float* proj_b      = (const float*)d_in[4];
  const float* rpb_table   = (const float*)d_in[5];
  const float* logit_scale = (const float*)d_in[6];
  const int*   rel_index   = (const int*)d_in[7];
  float* out = (float*)d_out;

  char* ws = (char*)d_ws;
  half_t* qkvw_h  = (half_t*)(ws + WS_QKVW);
  half_t* projw_h = (half_t*)(ws + WS_PROJW);
  float*  scales  = (float*)(ws + WS_SCALES);
  float*  biasc   = (float*)(ws + WS_BIASC);
  half_t* ao      = (half_t*)(ws + WS_AO);

  prep_kernel<<<4096, BLOCK, 0, stream>>>(qkv_w, proj_w, mask, rpb_table,
                                          rel_index, logit_scale, qkvw_h,
                                          projw_h, biasc, scales);

  swin_attn_kernel<<<NWIN, BLOCK, SMEM_BYTES, stream>>>(x, qkvw_h, biasc,
                                                        scales, ao);

  dim3 grid(NWIN, DIM / 64);
  proj_kernel<<<grid, BLOCK, 0, stream>>>(ao, projw_h, proj_b, out);
}
